// Mask_HybridMemory_81621558493655
// MI455X (gfx1250) — matrix operand order repaired and checked by reading, not compiled, not run on hardware
//
#include <hip/hip_runtime.h>
#include <hip/hip_bf16.h>

#define B_ 256
#define D_ 1024
#define N_ 65536
#define C_ 8192
#define INV_TEMP 20.0f
#define MOM 0.2f
#define EPS_ 1e-6f

typedef __attribute__((ext_vector_type(16))) __bf16 v16bf;
typedef __attribute__((ext_vector_type(8)))  float  v8f;

union Frag32B { v16bf v; uint4 q[2]; };

// ---- Pass 1: copy features f32 -> d_out (+1), convert to bf16 in ws ------
__global__ void prep_features(const float* __restrict__ feats,
                              float* __restrict__ out_feats,
                              __bf16* __restrict__ featbf) {
  size_t idx = (size_t)blockIdx.x * blockDim.x + threadIdx.x; // 16M threads, 4 elems each
  const float4 f4 = ((const float4*)feats)[idx];
  size_t e = idx * 4;
  out_feats[e + 0] = f4.x; out_feats[e + 1] = f4.y;
  out_feats[e + 2] = f4.z; out_feats[e + 3] = f4.w;
  union { __bf16 h[4]; uint2 u; } pk;
  pk.h[0] = (__bf16)f4.x; pk.h[1] = (__bf16)f4.y;
  pk.h[2] = (__bf16)f4.z; pk.h[3] = (__bf16)f4.w;
  ((uint2*)featbf)[idx] = pk.u;
}

// ---- zero sim accumulator + counts, convert inputs to bf16 ----------------
__global__ void misc_init(float* __restrict__ simsum, int* __restrict__ counts,
                          const float* __restrict__ inputs, __bf16* __restrict__ inpbf) {
  int i = blockIdx.x * blockDim.x + threadIdx.x;      // 2,097,152 threads
  simsum[i] = 0.0f;                                    // exactly C_*B_
  if (i < C_) counts[i] = 0;
  if (i < B_ * D_) inpbf[i] = (__bf16)inputs[i];
}

// ---- label histogram -------------------------------------------------------
__global__ void hist_kernel(const int* __restrict__ labels, int* __restrict__ counts) {
  int n = blockIdx.x * blockDim.x + threadIdx.x;
  atomicAdd(&counts[labels[n]], 1);
}

// ---- fused GEMM (bf16 WMMA) + segment-sum scatter into sim[C][B] ----------
// K-chunk = 64: A slab 256 x 72 bf16 (row = 9 uint4, 144B) -> 36 KB LDS.
// Lane m-stride = 36 dwords, gcd(36,64)=4 -> 16 distinct bank starts, no
// conflicts for ds_load_b128 / ds_store_b128.
__global__ __launch_bounds__(256) void gemm_scatter(
    const __bf16* __restrict__ inpbf, const __bf16* __restrict__ featbf,
    const int* __restrict__ labels, float* __restrict__ simsum) {
  __shared__ uint4 Aslab[256 * 9];

  const int tid = threadIdx.x;
  const int w  = tid >> 5;            // wave 0..7
  const int l  = tid & 31;            // lane
  const int lm = l & 15;
  const int hi = l >> 4;              // K-half select per ISA fragment layout
  const int nrow = blockIdx.x * 128 + w * 16 + lm;   // feature row for this lane
  const int c = labels[nrow];

  v8f acc[16] = {};                   // 16 M-tiles x 8 VGPRs = whole M=256 column

  const __bf16* brow = featbf + (size_t)nrow * D_ + hi * 8;

  auto ldA = [&](int mt, int s, Frag32B& f) {
    const int m = mt * 16 + lm;
    f.q[0] = Aslab[m * 9 + s * 4 + hi];
    f.q[1] = Aslab[m * 9 + s * 4 + 2 + hi];
  };

  for (int k = 0; k < D_; k += 64) {
    __syncthreads();                   // prior chunk's LDS reads complete
    { // stage A slab: thread t copies inputs row t, 128B
      const uint4* g = (const uint4*)(inpbf + (size_t)tid * D_ + k);
#pragma unroll
      for (int j = 0; j < 8; ++j) Aslab[tid * 9 + j] = g[j];
    }
    __syncthreads();

    // preload both subchunks' B fragments (one clause of global_load_b128)
    Frag32B bfr[2];
#pragma unroll
    for (int s = 0; s < 2; ++s) {
      const __bf16* p = brow + k + s * 32;
      bfr[s].q[0] = *(const uint4*)p;
      bfr[s].q[1] = *(const uint4*)(p + 16);
    }

#pragma unroll
    for (int s = 0; s < 2; ++s) {
      Frag32B cur, nxt;
      ldA(0, s, cur);
#pragma unroll
      for (int mt = 0; mt < 16; ++mt) {
        if (mt < 15) ldA(mt + 1, s, nxt);    // double-buffered A fragments
        acc[mt] = __builtin_amdgcn_wmma_f32_16x16x32_bf16(
            false, cur.v, false, bfr[s].v, (short)0, acc[mt], false, false);
        if (mt < 15) cur = nxt;
      }
    }
  }

  // scatter: lane owns column n=nrow; D tile rows b = mt*16 + r + hi*8
  float* base = simsum + (size_t)c * B_;
#pragma unroll
  for (int mt = 0; mt < 16; ++mt) {
#pragma unroll
    for (int r = 0; r < 8; ++r) {
      unsafeAtomicAdd(base + mt * 16 + r + hi * 8, acc[mt][r]);
    }
  }
}

// ---- masked softmax denominator + focal loss probe at target ---------------
__global__ void loss_kernel(const float* __restrict__ simsum, const int* __restrict__ counts,
                            const int* __restrict__ labels, const int* __restrict__ indexes,
                            float* __restrict__ floss) {
  const int b = blockIdx.x, tid = threadIdx.x;
  __shared__ float red[256];
  float s = 0.0f;
  for (int c = tid; c < C_; c += 256) {
    int cnt = counts[c];
    if (cnt > 0) {
      float v = simsum[(size_t)c * B_ + b] * (INV_TEMP / (float)cnt);
      s += __expf(v);
    }
  }
  red[tid] = s; __syncthreads();
  for (int st = 128; st > 0; st >>= 1) {
    if (tid < st) red[tid] += red[tid + st];
    __syncthreads();
  }
  if (tid == 0) {
    float denom = red[0] + EPS_;
    int tc  = labels[indexes[b]];
    int cnt = counts[tc];                          // >= 1 by construction
    float vt = simsum[(size_t)tc * B_ + b] * (INV_TEMP / (float)cnt);
    float pt = __expf(vt) / denom;
    float om = 1.0f - pt; om *= om; om *= om;      // (1-p)^4
    floss[b] = -om * __logf(pt + EPS_);
  }
}

// ---- prototype gather (class mean at target) + cosine terms ---------------
__global__ void proto_kernel(const float* __restrict__ feats, const float* __restrict__ another,
                             const float* __restrict__ inputs, const int* __restrict__ labels,
                             const int* __restrict__ indexes,
                             float* __restrict__ cml, float* __restrict__ cl) {
  const int b = blockIdx.x, tid = threadIdx.x;
  __shared__ int target, lcount;
  __shared__ int lidx[512];
  __shared__ float red[256];
  if (tid == 0) { target = labels[indexes[b]]; lcount = 0; }
  __syncthreads();
  const int tc = target;
  for (int n = tid; n < N_; n += 256) {
    if (labels[n] == tc) {
      int p = atomicAdd(&lcount, 1);
      if (p < 512) lidx[p] = n;
    }
  }
  __syncthreads();
  int cnt = lcount; if (cnt > 512) cnt = 512;
  float p4[4] = {0, 0, 0, 0};                    // proto sum; /count cancels in cosine
  for (int i = 0; i < cnt; ++i) {
    const float* row = feats + (size_t)lidx[i] * D_;
#pragma unroll
    for (int j = 0; j < 4; ++j) p4[j] += row[tid + j * 256];
  }
  float pp = 0, aa = 0, pa = 0, ii = 0, ia = 0;
#pragma unroll
  for (int j = 0; j < 4; ++j) {
    int d = tid + j * 256;
    float an = another[(size_t)b * D_ + d];
    float in = inputs[(size_t)b * D_ + d];
    pp += p4[j] * p4[j]; aa += an * an; pa += p4[j] * an;
    ii += in * in;       ia += in * an;
  }
  float vals[5] = {pp, aa, pa, ii, ia};
  float out5[5];
  for (int v = 0; v < 5; ++v) {
    red[tid] = vals[v]; __syncthreads();
    for (int st = 128; st > 0; st >>= 1) {
      if (tid < st) red[tid] += red[tid + st];
      __syncthreads();
    }
    out5[v] = red[0]; __syncthreads();
  }
  if (tid == 0) {
    cml[b] = out5[2] * rsqrtf(out5[0]) * rsqrtf(out5[1]);
    cl[b]  = out5[4] * rsqrtf(out5[3]) * rsqrtf(out5[1]);
  }
}

// ---- final scalar loss -----------------------------------------------------
__global__ void finalize_kernel(const float* __restrict__ floss, const float* __restrict__ cml,
                                const float* __restrict__ cl, const int* __restrict__ back,
                                float* __restrict__ out0) {
  const int tid = threadIdx.x;
  __shared__ float r1[256], r2[256], r3[256];
  r1[tid] = floss[tid]; r2[tid] = cml[tid]; r3[tid] = cl[tid];
  __syncthreads();
  for (int st = 128; st > 0; st >>= 1) {
    if (tid < st) { r1[tid] += r1[tid + st]; r2[tid] += r2[tid + st]; r3[tid] += r3[tid + st]; }
    __syncthreads();
  }
  if (tid == 0) {
    float loss = r1[0] / 256.0f;
    if (back[0] != 0) loss += -(r2[0] / 256.0f) - (r3[0] / 256.0f);
    out0[0] = loss;
  }
}

// ---- sequential EMA scatter with per-row renorm, chained by index ----------
__global__ void ema_kernel(const float* __restrict__ inputs, const float* __restrict__ msk,
                           const int* __restrict__ indexes, float* __restrict__ feat) {
  const int tid = threadIdx.x;                 // 1024 threads: 4 per b
  const int b = tid >> 2, q = tid & 3;
  __shared__ int sIdx[256], sPos[256], sMax;
  __shared__ float sRed[256][4];
  if (tid < 256) sIdx[tid] = indexes[tid];
  if (tid == 0) sMax = 0;
  __syncthreads();
  if (q == 0) {
    int y = sIdx[b], p = 0;
    for (int j = 0; j < b; ++j) p += (sIdx[j] == y);
    sPos[b] = p;
    atomicMax(&sMax, p);
  }
  __syncthreads();
  const int rounds = sMax;
  const size_t ib = (size_t)b * D_;
  for (int r = 0; r <= rounds; ++r) {
    const bool act = (sPos[b] == r);
    const size_t ro = (size_t)sIdx[b] * D_;
    float ssq = 0.0f;
    if (act) {
      for (int j = 0; j < 256; ++j) {
        int d = q * 256 + j;
        float v = MOM * feat[ro + d] + (1.0f - MOM) * (inputs[ib + d] + msk[ib + d]);
        ssq += v * v;
      }
    }
    sRed[b][q] = ssq;
    __syncthreads();
    float inv = rsqrtf(sRed[b][0] + sRed[b][1] + sRed[b][2] + sRed[b][3]);
    __syncthreads();
    if (act) {
      for (int j = 0; j < 256; ++j) {
        int d = q * 256 + j;
        float v = MOM * feat[ro + d] + (1.0f - MOM) * (inputs[ib + d] + msk[ib + d]);
        feat[ro + d] = v * inv;
      }
    }
    __threadfence();
    __syncthreads();
  }
}

extern "C" void kernel_launch(void* const* d_in, const int* in_sizes, int n_in,
                              void* d_out, int out_size, void* d_ws, size_t ws_size,
                              hipStream_t stream) {
  const float* inputs  = (const float*)d_in[0];
  const float* inmask  = (const float*)d_in[1];
  const float* another = (const float*)d_in[2];
  const float* feats   = (const float*)d_in[3];
  const int*   labels  = (const int*)d_in[5];
  const int*   indexes = (const int*)d_in[7];
  const int*   back    = (const int*)d_in[8];
  float* out = (float*)d_out;                  // [0]=loss, [1..]=new_features

  char* ws = (char*)d_ws;
  __bf16* featbf = (__bf16*)ws;  ws += (size_t)N_ * D_ * 2;   // 134 MB (L2-resident)
  __bf16* inpbf  = (__bf16*)ws;  ws += (size_t)B_ * D_ * 2;   // 512 KB
  float*  simsum = (float*)ws;   ws += (size_t)C_ * B_ * 4;   // 8 MB
  int*    counts = (int*)ws;     ws += (size_t)C_ * 4;        // 32 KB
  float*  floss  = (float*)ws;   ws += 256 * 4;
  float*  cml    = (float*)ws;   ws += 256 * 4;
  float*  cl     = (float*)ws;   ws += 256 * 4;

  prep_features<<<65536, 256, 0, stream>>>(feats, out + 1, featbf);
  misc_init<<<8192, 256, 0, stream>>>(simsum, counts, inputs, inpbf);
  hist_kernel<<<256, 256, 0, stream>>>(labels, counts);
  gemm_scatter<<<512, 256, 0, stream>>>(inpbf, featbf, labels, simsum);
  loss_kernel<<<256, 256, 0, stream>>>(simsum, counts, labels, indexes, floss);
  proto_kernel<<<256, 256, 0, stream>>>(feats, another, inputs, labels, indexes, cml, cl);
  finalize_kernel<<<1, 256, 0, stream>>>(floss, cml, cl, back, out);
  ema_kernel<<<1, 1024, 0, stream>>>(inputs, inmask, indexes, out + 1);
}
